// LSTMLM_58746562674801
// MI455X (gfx1250) — matrix operand order repaired and checked by reading, not compiled, not run on hardware
//
#include <hip/hip_runtime.h>
#include <hip/hip_bf16.h>
#include <math.h>

typedef __attribute__((ext_vector_type(16))) __bf16 v16bf;
typedef __attribute__((ext_vector_type(8)))  __bf16 v8bf;
typedef __attribute__((ext_vector_type(4)))  __bf16 v4bf;
typedef __attribute__((ext_vector_type(8)))  float  v8f;
typedef __attribute__((ext_vector_type(2)))  float  v2f;
typedef __attribute__((ext_vector_type(4)))  unsigned int v4u;
typedef __attribute__((ext_vector_type(8)))  int v8i;
typedef __attribute__((ext_vector_type(4)))  int v4i;

#define D_    1024
#define V_    32000
#define B_    8
#define T_    512
#define MROWS 4096      // B*T
#define LSTRIDE 40      // bf16 LDS row pitch = 32 data + 8 pad (TDM pads 4 DW / 16 DW)
#define TILE_E  (128 * LSTRIDE)
#define HS 1028         // f32 LDS row pitch for h
#define GS 132          // f32 LDS row pitch for gate staging

// ------------------------------------------------------------- f32 -> bf16 helpers
__global__ __launch_bounds__(256)
void conv_bf16_kernel(const float* __restrict__ src, __bf16* __restrict__ dst) {
  size_t i = (size_t)blockIdx.x * 256 + threadIdx.x;   // float4 index
  float4 v = ((const float4*)src)[i];
  v4bf o;
  o[0] = (__bf16)v.x; o[1] = (__bf16)v.y; o[2] = (__bf16)v.z; o[3] = (__bf16)v.w;
  *(v4bf*)(dst + 4 * i) = o;
}

__global__ __launch_bounds__(256)
void embed_bf16_kernel(const int* __restrict__ idx, const float* __restrict__ emb,
                       __bf16* __restrict__ x) {
  int row = blockIdx.x;
  int tok = idx[row];
  float4 v = ((const float4*)(emb + (size_t)tok * D_))[threadIdx.x];
  v4bf o;
  o[0] = (__bf16)v.x; o[1] = (__bf16)v.y; o[2] = (__bf16)v.z; o[3] = (__bf16)v.w;
  *(v4bf*)(x + (size_t)row * D_ + threadIdx.x * 4) = o;
}

__global__ __launch_bounds__(256)
void bias_add_kernel(const float* __restrict__ a, const float* __restrict__ b,
                     float* __restrict__ o) {
  int i = blockIdx.x * 256 + threadIdx.x;
  o[i] = a[i] + b[i];
}

// ------------------------------------------------------------- TDM tile fetch
// Loads a [rows=128, cols=32] bf16 tile (row-major, row stride K elements) from
// global into LDS at lds_off, padding 4 DWORDs every 16 DWORDs -> pitch 40 bf16.
__device__ __forceinline__ void tdm_load_tile(const __bf16* gptr, unsigned lds_off,
                                              int K, int rows_total)
{
  unsigned long long ga = (unsigned long long)(uintptr_t)gptr;
  v4u g0;
  g0[0] = 1u;                                           // count=1, user mode
  g0[1] = lds_off;                                      // LDS byte address
  g0[2] = (unsigned)(ga & 0xFFFFFFFFu);                 // global_addr[31:0]
  g0[3] = (unsigned)((ga >> 32) & 0x01FFFFFFu)          // global_addr[56:32]
          | 0x80000000u;                                // type=2 ("image")
  v8i g1;
  g1[0] = (1 << 16)        // data_size = 2 bytes
        | (1 << 20)        // pad_enable
        | (3 << 22)        // pad_interval: 16 DWORDs
        | (3 << 25);       // pad_amount:   4 DWORDs
  g1[1] = (K & 0xFFFF) << 16;                 // tensor_dim0[15:0]
  g1[2] = ((unsigned)K >> 16)                 // tensor_dim0[31:16]
        | ((rows_total & 0xFFFF) << 16);      // tensor_dim1[15:0]
  g1[3] = ((unsigned)rows_total >> 16)        // tensor_dim1[31:16]
        | (32 << 16);                         // tile_dim0 = 32
  g1[4] = 128;                                // tile_dim1 = 128, tile_dim2 = 0
  g1[5] = K;                                  // tensor_dim0_stride[31:0]
  g1[6] = 0;
  g1[7] = 0;
  v4i z4 = {};
  v8i z8 = {};
  // 6-arg form (this toolchain): (g0, g1, g2, g3, g4, cpol)
  __builtin_amdgcn_tensor_load_to_lds(g0, g1, z4, z4, z8, 0);
}

// ------------------------------------------------- bf16 WMMA GEMM: C = A*W^T + bias
// A: [M,K] bf16 row-major, W: [N,K] bf16 row-major, C: [M,N] f32, bias[N] f32.
// Block tile 128x128, K-step 32, 8 waves; wave owns 2x4 tiles of 16x16.
// LDS tiles are double-buffered and filled by the Tensor Data Mover (wave 0),
// overlapping DMA of tile k+1 with WMMA on tile k.
__global__ __launch_bounds__(256)
void gemm_bf16(const __bf16* __restrict__ A, const __bf16* __restrict__ W,
               const float* __restrict__ bias, float* __restrict__ C,
               int M, int N, int K)
{
  __shared__ __bf16 At[2 * TILE_E];
  __shared__ __bf16 Bt[2 * TILE_E];
  const int tid  = threadIdx.x;
  const int lane = tid & 31;
  const int wave = tid >> 5;
  const int wr = wave & 3;        // 32-row group
  const int wc = wave >> 2;       // 64-col group
  const int r    = lane & 15;
  const int half = lane >> 4;
  const int m0 = blockIdx.y * 128;
  const int n0 = blockIdx.x * 128;

  const unsigned ldsA0 = (unsigned)(uintptr_t)&At[0];
  const unsigned ldsA1 = (unsigned)(uintptr_t)&At[TILE_E];
  const unsigned ldsB0 = (unsigned)(uintptr_t)&Bt[0];
  const unsigned ldsB1 = (unsigned)(uintptr_t)&Bt[TILE_E];

  v8f acc[2][4] = {};

  const int nsteps = K / 32;
  if (wave == 0) {                         // prime buffer 0
    tdm_load_tile(A + (size_t)m0 * K, ldsA0, K, M);
    tdm_load_tile(W + (size_t)n0 * K, ldsB0, K, N);
  }

  for (int s = 0; s < nsteps; ++s) {
    const int k0  = s * 32;
    const int cur = s & 1;
    if (wave == 0) {
      if (s + 1 < nsteps) {                // prefetch next tile into other buffer
        tdm_load_tile(A + (size_t)m0 * K + (k0 + 32), cur ? ldsA0 : ldsA1, K, M);
        tdm_load_tile(W + (size_t)n0 * K + (k0 + 32), cur ? ldsB0 : ldsB1, K, N);
        __builtin_amdgcn_s_wait_tensorcnt(2);   // current buffer's 2 loads done
      } else {
        __builtin_amdgcn_s_wait_tensorcnt(0);
      }
    }
    __syncthreads();                       // LDS tiles visible to all waves

    const __bf16* Ab = &At[cur * TILE_E];
    const __bf16* Bb = &Bt[cur * TILE_E];

    v16bf af[2], bfr[4];
#pragma unroll
    for (int mt = 0; mt < 2; ++mt) {
      const __bf16* ap = &Ab[(wr * 32 + mt * 16 + r) * LSTRIDE + half * 8];
      v8bf lo = *(const v8bf*)ap;          // K = half*8 + 0..7
      v8bf hi = *(const v8bf*)(ap + 16);   // K = 16 + half*8 + 0..7
      v16bf a;
#pragma unroll
      for (int e = 0; e < 8; ++e) { a[e] = lo[e]; a[e + 8] = hi[e]; }
      af[mt] = a;
    }
#pragma unroll
    for (int nt = 0; nt < 4; ++nt) {
      const __bf16* bp = &Bb[(wc * 64 + nt * 16 + r) * LSTRIDE + half * 8];
      v8bf lo = *(const v8bf*)bp;          // K = half*8 + 0..7
      v8bf hi = *(const v8bf*)(bp + 16);   // K = 16 + half*8 + 0..7
      v16bf b;
#pragma unroll
      for (int e = 0; e < 8; ++e) { b[e] = lo[e]; b[e + 8] = hi[e]; }
      bfr[nt] = b;
    }
#pragma unroll
    for (int mt = 0; mt < 2; ++mt)
#pragma unroll
      for (int nt = 0; nt < 4; ++nt)
        acc[mt][nt] = __builtin_amdgcn_wmma_f32_16x16x32_bf16(
            false, af[mt], false, bfr[nt], (short)0, acc[mt][nt], false, false);
    __syncthreads();                       // done reading buffer before TDM reuses it
  }

  // epilogue: D layout = VGPR g -> row g (+8 for lanes 16..31), col = lane%16
  const int rofs = half * 8;
#pragma unroll
  for (int nt = 0; nt < 4; ++nt) {
    int col = n0 + wc * 64 + nt * 16 + r;
    float bv = bias[col];
#pragma unroll
    for (int mt = 0; mt < 2; ++mt) {
      int rowbase = m0 + wr * 32 + mt * 16 + rofs;
#pragma unroll
      for (int g = 0; g < 8; ++g)
        C[(size_t)(rowbase + g) * N + col] = acc[mt][nt][g] + bv;
    }
  }
}

// ---------------------------------------------------------------- LSTM recurrence
__device__ __forceinline__ float sigm(float x) { return 1.0f / (1.0f + expf(-x)); }

// Persistent kernel, 32 blocks * 256 threads. Block owns 32 hidden cols.
// Wave w -> gate q=w>>1, sub s=w&1 -> one 16x16 f32 WMMA tile of g = h @ W_hh^T.
// h double-buffered in global (hglob[2][8][D]); grid barrier per timestep.
__global__ __launch_bounds__(256)
void lstm_recur(const float* __restrict__ xg,   // [B*T, 4D] = x@W_ih^T + b_ih + b_hh
                const float* __restrict__ Whh,  // [4D, D] row-major
                float* __restrict__ hglob,      // [2][8][D]
                float* __restrict__ y,          // [B*T, D]
                unsigned* __restrict__ bar)
{
  __shared__ float hl[9 * HS];     // rows 0..7 = h, row 8 = zeros (batch pad 8->16)
  __shared__ float gl[16 * GS];    // gate tiles staging
  __shared__ float cl[8 * 32];     // cell state slice (persistent across steps)

  const int tid  = threadIdx.x;
  const int lane = tid & 31;
  const int wave = tid >> 5;
  const int q  = wave >> 1;
  const int sb = wave & 1;
  const int j0 = blockIdx.x * 32;
  const int n0 = q * D_ + j0 + sb * 16;
  const int r    = lane & 15;
  const int half = lane >> 4;
  const int G = gridDim.x;

  for (int i = tid; i < HS; i += 256) hl[8 * HS + i] = 0.0f;  // zero pad row
  cl[tid] = 0.0f;                                             // c0 = 0
  __syncthreads();

  const int rp = (r < 8) ? r : 8;                      // padded rows read zeros
  const float* wrow = Whh + (size_t)(n0 + r) * D_;     // B col n = n0 + lane%16

  for (int t = 0; t < T_; ++t) {
    const float* hin  = hglob + (size_t)(t & 1) * 8 * D_;
    float*       hout = hglob + (size_t)((t + 1) & 1) * 8 * D_;

    // load h_{t-1} (8 rows x 1024) into LDS
    for (int i = tid; i < 8 * 256; i += 256) {
      int row = i >> 8, seg = i & 255;
      *(float4*)(&hl[row * HS + seg * 4]) =
          *(const float4*)(hin + row * D_ + seg * 4);
    }
    __syncthreads();

    // g_tile[16,16] = h[16,1024] x Whh^T slice, via 256x wmma f32 16x16x4
    v8f acc = {};
    const float* arow = &hl[rp * HS];
    for (int k = 0; k < D_; k += 4) {
      v2f a = *(const v2f*)(arow + k + half * 2);   // A: K = (lane<16?0:2)+{0,1}
      v2f b = *(const v2f*)(wrow + k + half * 2);   // B: same K pair, col n
      acc = __builtin_amdgcn_wmma_f32_16x16x4_f32(
          false, a, false, b, (short)0, acc, false, false);
    }
#pragma unroll
    for (int g = 0; g < 8; ++g)
      gl[(g + half * 8) * GS + wave * 16 + r] = acc[g];
    __syncthreads();

    // elementwise gates: 8 batch x 32 hidden = 256 threads, one each
    {
      int b  = tid >> 5;
      int jj = tid & 31;
      int ss = jj >> 4, cw = jj & 15;
      const float* xr = xg + ((size_t)b * T_ + t) * (4 * D_) + j0;
      float gi = gl[b * GS + (0 * 2 + ss) * 16 + cw] + xr[0 * D_ + jj];
      float gf = gl[b * GS + (1 * 2 + ss) * 16 + cw] + xr[1 * D_ + jj];
      float gg = gl[b * GS + (2 * 2 + ss) * 16 + cw] + xr[2 * D_ + jj];
      float go = gl[b * GS + (3 * 2 + ss) * 16 + cw] + xr[3 * D_ + jj];
      float c = sigm(gf) * cl[tid] + sigm(gi) * tanhf(gg);
      float h = sigm(go) * tanhf(c);
      cl[tid] = c;
      hout[b * D_ + j0 + jj] = h;
      y[((size_t)b * T_ + t) * D_ + j0 + jj] = h;
    }

    // grid-wide barrier (monotone counter, phase = t)
    __threadfence();
    __syncthreads();
    if (tid == 0) {
      atomicAdd(bar, 1u);
      unsigned target = (unsigned)(G * (t + 1));
      while (__hip_atomic_load(bar, __ATOMIC_RELAXED, __HIP_MEMORY_SCOPE_AGENT) < target)
        __builtin_amdgcn_s_sleep(1);
    }
    __syncthreads();
    __threadfence();
  }
}

// ---------------------------------------------------------------- layernorm (in place)
__global__ __launch_bounds__(256)
void layernorm_kernel(float* __restrict__ x, const float* __restrict__ gamma,
                      const float* __restrict__ beta)
{
  __shared__ float red[256];
  const int tid = threadIdx.x;
  float* xr = x + (size_t)blockIdx.x * D_;
  float4 v = ((const float4*)xr)[tid];
  red[tid] = v.x + v.y + v.z + v.w;
  __syncthreads();
  for (int o = 128; o > 0; o >>= 1) { if (tid < o) red[tid] += red[tid + o]; __syncthreads(); }
  float mu = red[0] * (1.0f / D_);
  __syncthreads();
  float dx = v.x - mu, dy = v.y - mu, dz = v.z - mu, dw = v.w - mu;
  red[tid] = dx * dx + dy * dy + dz * dz + dw * dw;
  __syncthreads();
  for (int o = 128; o > 0; o >>= 1) { if (tid < o) red[tid] += red[tid + o]; __syncthreads(); }
  float inv = rsqrtf(red[0] * (1.0f / D_) + 1e-5f);
  int j = tid * 4;
  xr[j + 0] = dx * inv * gamma[j + 0] + beta[j + 0];
  xr[j + 1] = dy * inv * gamma[j + 1] + beta[j + 1];
  xr[j + 2] = dz * inv * gamma[j + 2] + beta[j + 2];
  xr[j + 3] = dw * inv * gamma[j + 3] + beta[j + 3];
}

// ---------------------------------------------------------------- loss
__global__ __launch_bounds__(256)
void nll_kernel(const float* __restrict__ logits, const int* __restrict__ targets,
                float* __restrict__ nll)
{
  __shared__ float red[256];
  const int tid = threadIdx.x;
  const float* lr = logits + (size_t)blockIdx.x * V_;
  float m = -3.402823466e38f;
  for (int i = tid; i < V_; i += 256) m = fmaxf(m, lr[i]);
  red[tid] = m; __syncthreads();
  for (int o = 128; o > 0; o >>= 1) { if (tid < o) red[tid] = fmaxf(red[tid], red[tid + o]); __syncthreads(); }
  m = red[0];
  __syncthreads();
  float s = 0.0f;
  for (int i = tid; i < V_; i += 256) s += expf(lr[i] - m);
  red[tid] = s; __syncthreads();
  for (int o = 128; o > 0; o >>= 1) { if (tid < o) red[tid] += red[tid + o]; __syncthreads(); }
  if (tid == 0) nll[blockIdx.x] = (m + logf(red[0])) - lr[targets[blockIdx.x]];
}

__global__ __launch_bounds__(256)
void loss_reduce_kernel(const float* __restrict__ nll, float* __restrict__ out)
{
  __shared__ float red[256];
  const int tid = threadIdx.x;
  float s = 0.0f;
  for (int i = tid; i < MROWS; i += 256) s += nll[i];
  red[tid] = s; __syncthreads();
  for (int o = 128; o > 0; o >>= 1) { if (tid < o) red[tid] += red[tid + o]; __syncthreads(); }
  if (tid == 0) *out = red[0] * (1.0f / MROWS);
}

__global__ void init_state_kernel(float* __restrict__ h, unsigned* __restrict__ bar)
{
  int i = blockIdx.x * blockDim.x + threadIdx.x;
  if (i < 2 * 8 * D_) h[i] = 0.0f;
  if (i == 0) *bar = 0u;
}

// ---------------------------------------------------------------- driver
extern "C" void kernel_launch(void* const* d_in, const int* in_sizes, int n_in,
                              void* d_out, int out_size, void* d_ws, size_t ws_size,
                              hipStream_t stream)
{
  (void)in_sizes; (void)n_in; (void)out_size; (void)ws_size;
  const int*   idx     = (const int*)  d_in[0];
  const int*   targets = (const int*)  d_in[1];
  const float* emb     = (const float*)d_in[2];
  const float* W_ih0   = (const float*)d_in[3];
  const float* W_hh0   = (const float*)d_in[4];
  const float* b_ih0   = (const float*)d_in[5];
  const float* b_hh0   = (const float*)d_in[6];
  const float* W_ih1   = (const float*)d_in[7];
  const float* W_hh1   = (const float*)d_in[8];
  const float* b_ih1   = (const float*)d_in[9];
  const float* b_hh1   = (const float*)d_in[10];
  const float* gamma   = (const float*)d_in[11];
  const float* beta    = (const float*)d_in[12];
  const float* W_head  = (const float*)d_in[13];
  const float* b_head  = (const float*)d_in[14];

  float* logits = (float*)d_out;
  float* loss   = logits + (size_t)MROWS * V_;

  // workspace carve-up (~112 MB)
  float*    xB    = (float*)d_ws;                         // [4096,1024] f32
  float*    xA    = xB + (size_t)MROWS * D_;              // [4096,1024] f32
  float*    xg    = xA + (size_t)MROWS * D_;              // [4096,4096] f32 (64 MB)
  __bf16*   Abf   = (__bf16*)(xg + (size_t)MROWS * 4 * D_);   // [4096,1024] bf16
  __bf16*   Wibf  = Abf + (size_t)MROWS * D_;                 // [4096,1024] bf16
  float*    hgl   = (float*)(Wibf + (size_t)4 * D_ * D_);     // [2][8][1024]
  unsigned* bar   = (unsigned*)(hgl + 2 * 8 * D_);
  float*    nllb  = (float*)(bar + 16);                       // [4096]
  float*    biasc = nllb + MROWS;                             // [4096]
  __bf16*   Whbf  = (__bf16*)xg;   // head weight bf16 reuses xg after recurrences

  // ---- layer 0
  embed_bf16_kernel<<<MROWS, 256, 0, stream>>>(idx, emb, Abf);
  conv_bf16_kernel<<<4 * D_ * D_ / 1024, 256, 0, stream>>>(W_ih0, Wibf);
  bias_add_kernel<<<4 * D_ / 256, 256, 0, stream>>>(b_ih0, b_hh0, biasc);
  init_state_kernel<<<64, 256, 0, stream>>>(hgl, bar);
  gemm_bf16<<<dim3(4 * D_ / 128, MROWS / 128), 256, 0, stream>>>(
      Abf, Wibf, biasc, xg, MROWS, 4 * D_, D_);
  lstm_recur<<<32, 256, 0, stream>>>(xg, W_hh0, hgl, xB, bar);

  // ---- layer 1
  conv_bf16_kernel<<<MROWS * D_ / 1024, 256, 0, stream>>>(xB, Abf);
  conv_bf16_kernel<<<4 * D_ * D_ / 1024, 256, 0, stream>>>(W_ih1, Wibf);
  bias_add_kernel<<<4 * D_ / 256, 256, 0, stream>>>(b_ih1, b_hh1, biasc);
  init_state_kernel<<<64, 256, 0, stream>>>(hgl, bar);
  gemm_bf16<<<dim3(4 * D_ / 128, MROWS / 128), 256, 0, stream>>>(
      Abf, Wibf, biasc, xg, MROWS, 4 * D_, D_);
  lstm_recur<<<32, 256, 0, stream>>>(xg, W_hh1, hgl, xA, bar);

  // ---- layernorm + head + loss
  layernorm_kernel<<<MROWS, 256, 0, stream>>>(xA, gamma, beta);
  conv_bf16_kernel<<<MROWS * D_ / 1024, 256, 0, stream>>>(xA, Abf);
  conv_bf16_kernel<<<V_ * (D_ / 1024), 256, 0, stream>>>(W_head, Whbf); // 32000 blocks
  gemm_bf16<<<dim3(V_ / 128, MROWS / 128), 256, 0, stream>>>(
      Abf, Whbf, b_head, logits, MROWS, V_, D_);
  nll_kernel<<<MROWS, 256, 0, stream>>>(logits, targets, nllb);
  loss_reduce_kernel<<<1, 256, 0, stream>>>(nllb, loss);
}
